// MultiHead_67164698575161
// MI455X (gfx1250) — matrix operand order repaired and checked by reading, not compiled, not run on hardware
//
#include <hip/hip_runtime.h>
#include <stdint.h>

typedef __attribute__((ext_vector_type(16))) __bf16 v16bf;
typedef __attribute__((ext_vector_type(8)))  float  v8f;
typedef __attribute__((ext_vector_type(4)))  unsigned int v4u;
typedef __attribute__((ext_vector_type(8)))  int v8i;
typedef __attribute__((ext_vector_type(4)))  int v4i;

#define B_  2
#define L_  2048
#define D_  1024
#define H_  16
#define DH_ 64
#define M_  (B_ * L_)   // 4096 rows total

// ---- WMMA fragment index maps (CDNA5 ISA 7.12.2, wave32) ----------------
// A-matrix 16x32 bf16: lane half holds K=hl*16 .. hl*16+15 (contiguous).
__device__ __forceinline__ int kmapA(int e, int hl) {
  return hl * 16 + e;
}
// B-matrix 32x16 bf16: lane half holds K=hl*16 .. hl*16+15 (contiguous).

__device__ __forceinline__ v8f wmma_bf16(v16bf a, v16bf b, v8f c) {
  return __builtin_amdgcn_wmma_f32_16x16x32_bf16(false, a, false, b,
                                                 (short)0, c, false, false);
}
__device__ __forceinline__ __bf16 to_bf(float x) { return (__bf16)x; }

// ---- CDNA5 async global->LDS copy (16B/lane, ASYNCcnt-tracked) ----------
__device__ __forceinline__ unsigned lds_off(const void* p) {
  return (unsigned)(uintptr_t)p;   // low 32 bits of generic ptr == LDS offset
}
__device__ __forceinline__ void async_cp16(unsigned lds, const void* g) {
  asm volatile("global_load_async_to_lds_b128 %0, %1, off"
               :: "v"(lds), "v"(g) : "memory");
}
__device__ __forceinline__ void wait_async0() {
  asm volatile("s_wait_asynccnt 0x0" ::: "memory");
}

// ---- CDNA5 Tensor Data Mover: 64x64 bf16 2D tile -> LDS (padded rows) ---
// D# per ISA 8.3-8.6: group0 = {ctrl, lds_addr, global_addr, type=2},
// group1 = {flags/pad, tensor dims, tile dims, strides}; 2D => g2/g3 = 0.
// LDS pad: +16B after every 128B row (pad_interval=4 -> 32 DWORDs,
// pad_amount=3 -> 4 DWORDs), matching the [64][72] bf16 LDS layout.
// clang-23 toolchain: 6-arg builtin (g0 v4u, g1 v8i, v4i, v4i, v8i, cpol).
__device__ __forceinline__ void tdm_load_64x64_bf16(
    const void* gsrc, unsigned lds_byte, int row_stride_elems)
{
  unsigned long long ga = (unsigned long long)(uintptr_t)gsrc;
  v4u g0;
  g0[0] = 1u;                                        // count=1, user D#
  g0[1] = lds_byte;                                  // lds_addr
  g0[2] = (unsigned)(ga & 0xFFFFFFFFu);              // global_addr[31:0]
  g0[3] = (unsigned)((ga >> 32) & 0x01FFFFFFu) | 0x80000000u; // [56:32]|type=2
  v8i g1;
  g1[0] = (1 << 16) | (1 << 20) | (4 << 22) | (3 << 25); // 2B elems, pad cfg
  g1[1] = (64 << 16);                                // tensor_dim0 = 64
  g1[2] = (64 << 16);                                // tensor_dim1 = 64
  g1[3] = (64 << 16);                                // tile_dim0   = 64
  g1[4] = 64;                                        // tile_dim1   = 64
  g1[5] = row_stride_elems;                          // tensor_dim0_stride
  g1[6] = 0;
  g1[7] = 0;
  v4i g2 = {0, 0, 0, 0};
  v4i g3 = {0, 0, 0, 0};
  v8i g4 = {0, 0, 0, 0, 0, 0, 0, 0};
  __builtin_amdgcn_tensor_load_to_lds(g0, g1, g2, g3, g4, 0);
}

// =========================================================================
// Kernel 1: projection GEMM (fp32 in, bf16 WMMA, bf16 head-major out).
// Double-buffered LDS; fp32->bf16 staging pipelined through VGPRs.
// =========================================================================
__global__ __launch_bounds__(256) void proj_gemm_kernel(
    const float* __restrict__ A, const float* __restrict__ W,
    const float* __restrict__ bias, __bf16* __restrict__ outHM)
{
  __shared__ __bf16 As [2][128][40];  // [buf][m][k]  k-contiguous
  __shared__ __bf16 Bst[2][128][40];  // [buf][n][k]  k-contiguous (transposed W)

  const int tid  = threadIdx.x;
  const int wave = tid >> 5;
  const int lane = tid & 31;
  const int hl   = lane >> 4;
  const int ln   = lane & 15;
  const int m0   = blockIdx.x * 128;
  const int n0   = blockIdx.y * 128;
  const int wm   = (wave >> 1) * 32;
  const int wn   = (wave & 1) * 64;

  const int ar  = tid >> 1,  ac0 = (tid & 1) * 16;   // A: row, k-range
  const int bn  = tid & 127, bk0 = (tid >> 7) * 16;  // W: col, k-range

  // ---- prologue: stage K-tile 0 into buffer 0 ----
  {
    float areg[16], breg[16];
    #pragma unroll
    for (int i = 0; i < 16; ++i) areg[i] = A[(size_t)(m0 + ar) * D_ + ac0 + i];
    #pragma unroll
    for (int i = 0; i < 16; ++i) breg[i] = W[(size_t)(bk0 + i) * D_ + n0 + bn];
    #pragma unroll
    for (int i = 0; i < 16; ++i) As [0][ar][ac0 + i] = to_bf(areg[i]);
    #pragma unroll
    for (int i = 0; i < 16; ++i) Bst[0][bn][bk0 + i] = to_bf(breg[i]);
  }
  __syncthreads();

  v8f acc[2][4];
  const v8f vzero = {0.f,0.f,0.f,0.f,0.f,0.f,0.f,0.f};
  #pragma unroll
  for (int i = 0; i < 2; ++i)
    #pragma unroll
    for (int j = 0; j < 4; ++j) acc[i][j] = vzero;

  for (int kk = 0; kk < D_; kk += 32) {
    const int  buf  = (kk >> 5) & 1;
    const int  nbuf = buf ^ 1;
    const bool nxt  = (kk + 32 < D_);

    float areg[16], breg[16];
    if (nxt) {
      #pragma unroll
      for (int i = 0; i < 16; ++i)
        areg[i] = A[(size_t)(m0 + ar) * D_ + kk + 32 + ac0 + i];
      #pragma unroll
      for (int i = 0; i < 16; ++i)
        breg[i] = W[(size_t)(kk + 32 + bk0 + i) * D_ + n0 + bn];
    }

    v16bf af[2];
    #pragma unroll
    for (int mt = 0; mt < 2; ++mt) {
      int row = wm + mt * 16 + ln;
      #pragma unroll
      for (int e = 0; e < 16; ++e) af[mt][e] = As[buf][row][kmapA(e, hl)];
    }
    v16bf bfr[4];
    #pragma unroll
    for (int nt = 0; nt < 4; ++nt) {
      int col = wn + nt * 16 + ln;
      #pragma unroll
      for (int e = 0; e < 16; ++e) bfr[nt][e] = Bst[buf][col][hl * 16 + e];
    }
    #pragma unroll
    for (int mt = 0; mt < 2; ++mt)
      #pragma unroll
      for (int nt = 0; nt < 4; ++nt)
        acc[mt][nt] = wmma_bf16(af[mt], bfr[nt], acc[mt][nt]);

    if (nxt) {
      #pragma unroll
      for (int i = 0; i < 16; ++i) As [nbuf][ar][ac0 + i] = to_bf(areg[i]);
      #pragma unroll
      for (int i = 0; i < 16; ++i) Bst[nbuf][bn][bk0 + i] = to_bf(breg[i]);
    }
    __syncthreads();
  }

  // epilogue: bias add, scatter into head-major bf16 [B,H,L,DH]
  #pragma unroll
  for (int mt = 0; mt < 2; ++mt) {
    int mbase = m0 + wm + mt * 16 + 8 * hl;
    #pragma unroll
    for (int nt = 0; nt < 4; ++nt) {
      int n = n0 + wn + nt * 16 + ln;
      float bb = bias[n];
      int h = n >> 6, dh = n & 63;
      #pragma unroll
      for (int v = 0; v < 8; ++v) {
        int mm = mbase + v;
        int bidx = mm >> 11, ll = mm & (L_ - 1);
        size_t off = ((size_t)((bidx * H_ + h) * L_ + ll)) * DH_ + dh;
        outHM[off] = to_bf(acc[mt][nt][v] + bb);
      }
    }
  }
}

// =========================================================================
// Kernel 2: flash attention.  grid = (L/128, B*H); wave owns 16 query rows.
// K-tile: Tensor Data Mover (one descriptor per 64-key block, wave 0 issues,
// TENSORcnt-synced).  V-tile: transposed [dh][key] through VGPRs.
// Double-buffered; one barrier per 64-key block.
// =========================================================================
__global__ __launch_bounds__(256) void attn_kernel(
    const __bf16* __restrict__ Q, const __bf16* __restrict__ K,
    const __bf16* __restrict__ V, const unsigned char* __restrict__ mask,
    __bf16* __restrict__ ctx)
{
  __shared__ __bf16 Ks[2][64][72];   // [buf][key][dh]  (TDM-filled, padded)
  __shared__ __bf16 Vt[2][64][72];   // [buf][dh][key]  (transposed)
  __shared__ __bf16 Ps[8][16][72];   // per-wave P staging

  const int tid  = threadIdx.x;
  const int wave = tid >> 5;
  const int lane = tid & 31;
  const int hl   = lane >> 4;
  const int ln   = lane & 15;
  const int bh   = blockIdx.y;           // b*H + h
  const int b    = bh >> 4;
  const int h    = bh & (H_ - 1);
  const int rw   = blockIdx.x * 128 + wave * 16;

  const __bf16* Qh = Q + (size_t)bh * L_ * DH_;
  const __bf16* Kh = K + (size_t)bh * L_ * DH_;
  const __bf16* Vh = V + (size_t)bh * L_ * DH_;

  const int vdh = tid & 63, vk0 = (tid >> 6) * 16;   // V transpose: dh, keys

  // ---- Q fragments resident in VGPRs for the whole kernel ----
  v16bf aq[2];
  #pragma unroll
  for (int c = 0; c < 2; ++c)
    #pragma unroll
    for (int e = 0; e < 16; ++e)
      aq[c][e] = Qh[(size_t)(rw + ln) * DH_ + c * 32 + kmapA(e, hl)];

  // ---- prologue: stage key-block 0 into buffer 0 ----
  if (wave == 0)
    tdm_load_64x64_bf16(Kh, lds_off(&Ks[0][0][0]), DH_);
  {
    __bf16 vr[16];
    #pragma unroll
    for (int i = 0; i < 16; ++i) vr[i] = Vh[(size_t)(vk0 + i) * DH_ + vdh];
    #pragma unroll
    for (int i = 0; i < 16; ++i) Vt[0][vdh][vk0 + i] = vr[i];
  }
  if (wave == 0) __builtin_amdgcn_s_wait_tensorcnt(0);
  __syncthreads();

  const v8f vzero = {0.f,0.f,0.f,0.f,0.f,0.f,0.f,0.f};
  v8f acc[4];
  #pragma unroll
  for (int ct = 0; ct < 4; ++ct) acc[ct] = vzero;
  float rm[8], rs[8];
  #pragma unroll
  for (int v = 0; v < 8; ++v) { rm[v] = -1e30f; rs[v] = 0.f; }

  const float sc = 0.03125f;  // 1/sqrt(d_model) = 1/32
  const unsigned char* mrow = mask + (size_t)b * L_ * L_;

  for (int mb = 0; mb < L_; mb += 64) {
    const int  buf  = (mb >> 6) & 1;
    const int  nbuf = buf ^ 1;
    const bool nxt  = (mb + 64 < L_);

    // issue next K-block TDM + next V loads (overlap with compute)
    if (nxt && wave == 0)
      tdm_load_64x64_bf16(Kh + (size_t)(mb + 64) * DH_,
                          lds_off(&Ks[nbuf][0][0]), DH_);
    __bf16 vr[16];
    if (nxt) {
      #pragma unroll
      for (int i = 0; i < 16; ++i)
        vr[i] = Vh[(size_t)(mb + 64 + vk0 + i) * DH_ + vdh];
    }

    // ---- S = Q * K^T (b-frags contiguous in Ks) ----
    v8f s[4];
    #pragma unroll
    for (int nt = 0; nt < 4; ++nt) {
      v16bf bk0v, bk1v;
      #pragma unroll
      for (int e = 0; e < 16; ++e) {
        bk0v[e] = Ks[buf][nt * 16 + ln][hl * 16 + e];
        bk1v[e] = Ks[buf][nt * 16 + ln][32 + hl * 16 + e];
      }
      s[nt] = wmma_bf16(aq[0], bk0v, vzero);
      s[nt] = wmma_bf16(aq[1], bk1v, s[nt]);
    }

    // ---- scale + mask ----
    #pragma unroll
    for (int nt = 0; nt < 4; ++nt) {
      int col = mb + nt * 16 + ln;
      #pragma unroll
      for (int v = 0; v < 8; ++v) {
        int row = rw + v + 8 * hl;
        float x = s[nt][v] * sc;
        if (mrow[(size_t)row * L_ + col]) x = -1e30f;
        s[nt][v] = x;
      }
    }

    // ---- online softmax (16-lane halves hold distinct row sets) ----
    #pragma unroll
    for (int v = 0; v < 8; ++v) {
      float nm = s[0][v];
      #pragma unroll
      for (int nt = 1; nt < 4; ++nt) nm = fmaxf(nm, s[nt][v]);
      #pragma unroll
      for (int off = 8; off >= 1; off >>= 1) nm = fmaxf(nm, __shfl_xor(nm, off));
      float newm = fmaxf(rm[v], nm);
      float corr = __expf(rm[v] - newm);
      rm[v] = newm;
      float ps = 0.f;
      #pragma unroll
      for (int nt = 0; nt < 4; ++nt) {
        float p = __expf(s[nt][v] - newm);
        s[nt][v] = p;
        ps += p;
      }
      #pragma unroll
      for (int off = 8; off >= 1; off >>= 1) ps += __shfl_xor(ps, off);
      rs[v] = rs[v] * corr + ps;
      #pragma unroll
      for (int ct = 0; ct < 4; ++ct) acc[ct][v] *= corr;
    }

    // ---- re-fragment P via wave-private LDS, ctx += P * V ----
    #pragma unroll
    for (int nt = 0; nt < 4; ++nt)
      #pragma unroll
      for (int v = 0; v < 8; ++v)
        Ps[wave][v + 8 * hl][nt * 16 + ln] = to_bf(s[nt][v]);

    v16bf pa[2];
    #pragma unroll
    for (int c = 0; c < 2; ++c)
      #pragma unroll
      for (int e = 0; e < 16; ++e)
        pa[c][e] = Ps[wave][ln][c * 32 + kmapA(e, hl)];

    #pragma unroll
    for (int ct = 0; ct < 4; ++ct) {
      v16bf bv0, bv1;
      #pragma unroll
      for (int e = 0; e < 16; ++e) {
        bv0[e] = Vt[buf][ct * 16 + ln][hl * 16 + e];
        bv1[e] = Vt[buf][ct * 16 + ln][32 + hl * 16 + e];
      }
      acc[ct] = wmma_bf16(pa[0], bv0, acc[ct]);
      acc[ct] = wmma_bf16(pa[1], bv1, acc[ct]);
    }

    // complete next-tile staging, then block-wide barrier
    if (nxt) {
      #pragma unroll
      for (int i = 0; i < 16; ++i) Vt[nbuf][vdh][vk0 + i] = vr[i];
    }
    if (wave == 0) __builtin_amdgcn_s_wait_tensorcnt(0);
    __syncthreads();
  }

  // ---- normalize, write ctx as bf16 row-major [B*L, D] ----
  #pragma unroll
  for (int v = 0; v < 8; ++v) {
    float inv = 1.f / rs[v];
    int row = rw + v + 8 * hl;
    size_t base = ((size_t)(b * L_ + row)) * D_ + h * DH_;
    #pragma unroll
    for (int ct = 0; ct < 4; ++ct)
      ctx[base + ct * 16 + ln] = to_bf(acc[ct][v] * inv);
  }
}

// =========================================================================
// Kernel 3: FC GEMM: out = values + relu(ctx * Wfc + bfc)  (pre-LayerNorm)
// ctx A-tiles: async global->LDS (already bf16).  Wfc: VGPR-pipelined cvt.
// =========================================================================
__global__ __launch_bounds__(256) void fc_gemm_kernel(
    const __bf16* __restrict__ Actx, const float* __restrict__ W,
    const float* __restrict__ bias, const float* __restrict__ resid,
    float* __restrict__ out)
{
  __shared__ __bf16 As [2][128][40];  // [buf][m][k]
  __shared__ __bf16 Bst[2][128][40];  // [buf][n][k]

  const int tid  = threadIdx.x;
  const int wave = tid >> 5;
  const int lane = tid & 31;
  const int hl   = lane >> 4;
  const int ln   = lane & 15;
  const int m0   = blockIdx.x * 128;
  const int n0   = blockIdx.y * 128;
  const int wm   = (wave >> 1) * 32;
  const int wn   = (wave & 1) * 64;

  const int bn = tid & 127, bk0 = (tid >> 7) * 16;

  // ---- prologue ----
  #pragma unroll
  for (int j = 0; j < 2; ++j) {
    int ch = tid + 256 * j;
    int r = ch >> 2, c8 = (ch & 3) * 8;
    async_cp16(lds_off(&As[0][r][c8]), &Actx[(size_t)(m0 + r) * D_ + c8]);
  }
  {
    float breg[16];
    #pragma unroll
    for (int i = 0; i < 16; ++i) breg[i] = W[(size_t)(bk0 + i) * D_ + n0 + bn];
    #pragma unroll
    for (int i = 0; i < 16; ++i) Bst[0][bn][bk0 + i] = to_bf(breg[i]);
  }
  wait_async0();
  __syncthreads();

  v8f acc[2][4];
  const v8f vzero = {0.f,0.f,0.f,0.f,0.f,0.f,0.f,0.f};
  #pragma unroll
  for (int i = 0; i < 2; ++i)
    #pragma unroll
    for (int j = 0; j < 4; ++j) acc[i][j] = vzero;

  for (int kk = 0; kk < D_; kk += 32) {
    const int  buf  = (kk >> 5) & 1;
    const int  nbuf = buf ^ 1;
    const bool nxt  = (kk + 32 < D_);

    float breg[16];
    if (nxt) {
      #pragma unroll
      for (int j = 0; j < 2; ++j) {
        int ch = tid + 256 * j;
        int r = ch >> 2, c8 = (ch & 3) * 8;
        async_cp16(lds_off(&As[nbuf][r][c8]),
                   &Actx[(size_t)(m0 + r) * D_ + kk + 32 + c8]);
      }
      #pragma unroll
      for (int i = 0; i < 16; ++i)
        breg[i] = W[(size_t)(kk + 32 + bk0 + i) * D_ + n0 + bn];
    }

    v16bf af[2];
    #pragma unroll
    for (int mt = 0; mt < 2; ++mt) {
      int row = wm + mt * 16 + ln;
      #pragma unroll
      for (int e = 0; e < 16; ++e) af[mt][e] = As[buf][row][kmapA(e, hl)];
    }
    v16bf bfr[4];
    #pragma unroll
    for (int nt = 0; nt < 4; ++nt) {
      int col = wn + nt * 16 + ln;
      #pragma unroll
      for (int e = 0; e < 16; ++e) bfr[nt][e] = Bst[buf][col][hl * 16 + e];
    }
    #pragma unroll
    for (int mt = 0; mt < 2; ++mt)
      #pragma unroll
      for (int nt = 0; nt < 4; ++nt)
        acc[mt][nt] = wmma_bf16(af[mt], bfr[nt], acc[mt][nt]);

    if (nxt) {
      #pragma unroll
      for (int i = 0; i < 16; ++i) Bst[nbuf][bn][bk0 + i] = to_bf(breg[i]);
    }
    wait_async0();
    __syncthreads();
  }

  #pragma unroll
  for (int mt = 0; mt < 2; ++mt) {
    int mbase = m0 + wm + mt * 16 + 8 * hl;
    #pragma unroll
    for (int nt = 0; nt < 4; ++nt) {
      int n = n0 + wn + nt * 16 + ln;
      float bb = bias[n];
      #pragma unroll
      for (int v = 0; v < 8; ++v) {
        int mm = mbase + v;
        float x = fmaxf(acc[mt][nt][v] + bb, 0.f);      // relu
        size_t off = (size_t)mm * D_ + n;
        out[off] = resid[off] + x;                      // + values residual
      }
    }
  }
}

// =========================================================================
// Kernel 4: row-wise LayerNorm, in place on d_out.  One block per row.
// =========================================================================
__global__ __launch_bounds__(256) void ln_kernel(
    float* __restrict__ y, const float* __restrict__ gamma,
    const float* __restrict__ beta)
{
  __shared__ float sh1[8];
  __shared__ float sh2[8];
  const int tid = threadIdx.x;
  float* p = y + (size_t)blockIdx.x * D_;

  float x[4];
  float s = 0.f;
  #pragma unroll
  for (int i = 0; i < 4; ++i) { x[i] = p[tid + 256 * i]; s += x[i]; }
  #pragma unroll
  for (int off = 16; off >= 1; off >>= 1) s += __shfl_xor(s, off);
  if ((tid & 31) == 0) sh1[tid >> 5] = s;
  __syncthreads();
  float tot = 0.f;
  #pragma unroll
  for (int i = 0; i < 8; ++i) tot += sh1[i];
  float mu = tot * (1.f / D_);

  float s2 = 0.f;
  #pragma unroll
  for (int i = 0; i < 4; ++i) { float d = x[i] - mu; s2 += d * d; }
  #pragma unroll
  for (int off = 16; off >= 1; off >>= 1) s2 += __shfl_xor(s2, off);
  if ((tid & 31) == 0) sh2[tid >> 5] = s2;
  __syncthreads();
  float tot2 = 0.f;
  #pragma unroll
  for (int i = 0; i < 8; ++i) tot2 += sh2[i];
  float inv = rsqrtf(tot2 * (1.f / D_) + 1e-5f);

  #pragma unroll
  for (int i = 0; i < 4; ++i) {
    int j = tid + 256 * i;
    p[j] = (x[i] - mu) * inv * gamma[j] + beta[j];
  }
}

// =========================================================================
extern "C" void kernel_launch(void* const* d_in, const int* in_sizes, int n_in,
                              void* d_out, int out_size, void* d_ws, size_t ws_size,
                              hipStream_t stream) {
  (void)in_sizes; (void)n_in; (void)out_size; (void)ws_size;
  const float* queries = (const float*)d_in[0];
  const float* keys    = (const float*)d_in[1];
  const float* values  = (const float*)d_in[2];
  const unsigned char* mask = (const unsigned char*)d_in[3];
  const float* Wq  = (const float*)d_in[4];
  const float* bq  = (const float*)d_in[5];
  const float* Wk  = (const float*)d_in[6];
  const float* bk  = (const float*)d_in[7];
  const float* Wv  = (const float*)d_in[8];
  const float* bv  = (const float*)d_in[9];
  const float* Wfc = (const float*)d_in[10];
  const float* bfc = (const float*)d_in[11];
  const float* gamma = (const float*)d_in[12];
  const float* beta  = (const float*)d_in[13];
  float* out = (float*)d_out;

  const size_t NE = (size_t)M_ * D_;          // 4,194,304 elements
  __bf16* Qbf = (__bf16*)d_ws;                // [B,H,L,DH] bf16
  __bf16* Kbf = Qbf + NE;
  __bf16* Vbf = Kbf + NE;
  __bf16* Ctx = Vbf + NE;                     // [B*L, D] bf16

  dim3 gg(M_ / 128, D_ / 128);                // (32, 8)
  proj_gemm_kernel<<<gg, 256, 0, stream>>>(queries, Wq, bq, Qbf);
  proj_gemm_kernel<<<gg, 256, 0, stream>>>(keys,    Wk, bk, Kbf);
  proj_gemm_kernel<<<gg, 256, 0, stream>>>(values,  Wv, bv, Vbf);

  attn_kernel<<<dim3(L_ / 128, B_ * H_), 256, 0, stream>>>(Qbf, Kbf, Vbf, mask, Ctx);

  fc_gemm_kernel<<<gg, 256, 0, stream>>>(Ctx, Wfc, bfc, values, out);

  ln_kernel<<<M_, 256, 0, stream>>>(out, gamma, beta);
}
